// TransformerBlock_20985210208376
// MI455X (gfx1250) — matrix operand order repaired and checked by reading, not compiled, not run on hardware
//
#include <hip/hip_runtime.h>
#include <hip/hip_bf16.h>
#include <math.h>

// ---------------------------------------------------------------------------
// Transformer block for MI455X (gfx1250, wave32, WMMA, async-LDS).
// Compute-bound (~275 GFLOP, AI ~700 flop/B at 23.3 TB/s): every GEMM runs
// on v_wmma_f32_16x16x32_bf16. All matrix operands are pre-converted to bf16
// once, so the hot loops are pure {async global->LDS copy, ds_load, wmma}.
// GEMM K-tiles and attention K/V chunks are double-buffered with
// GLOBAL_LOAD_ASYNC_TO_LDS_B128 (ASYNCcnt) when the builtin is available.
// ---------------------------------------------------------------------------

#define D_MODEL   1024
#define N_HEADS   16
#define HEAD_DIM  64
#define FF_DIM    4096
#define SEQ       2048
#define BATCH     4
#define ROWS      (BATCH * SEQ)        // 8192 token rows

typedef __attribute__((ext_vector_type(16))) __bf16 v16bf;
typedef __attribute__((ext_vector_type(8)))  float  v8f;
typedef int v4i __attribute__((vector_size(16)));   // matches async-LDS builtin

union Frag { v16bf v; unsigned int u[8]; unsigned short s[16]; };

#ifndef __has_builtin
#define __has_builtin(x) 0
#endif
#if __has_builtin(__builtin_amdgcn_global_load_async_to_lds_b128)
#define ASYNC_LDS 1
#endif

__device__ __forceinline__ unsigned short f2bf(float f) {
  unsigned int u = __float_as_uint(f);
  u += 0x7FFFu + ((u >> 16) & 1u);          // round-to-nearest-even
  return (unsigned short)(u >> 16);
}
__device__ __forceinline__ unsigned int packbf2(float a, float b) {
  return (unsigned int)f2bf(a) | ((unsigned int)f2bf(b) << 16);
}
// 16-bit A-matrix 16x32 lane layout (contiguous K order, matches B): VGPR j -> K index
__device__ __forceinline__ int a_kd(int j, int lhalf) {
  return 16 * lhalf + 2 * j;
}

// 16-byte global->LDS copy: async (ASYNCcnt) if available, else via VGPRs.
__device__ __forceinline__ void cp_b128(const unsigned short* g, unsigned short* l) {
#ifdef ASYNC_LDS
  __builtin_amdgcn_global_load_async_to_lds_b128((v4i*)g, (v4i*)l, 0, 0);
#else
  *reinterpret_cast<uint4*>(l) = *reinterpret_cast<const uint4*>(g);
#endif
}

template <int N>
__device__ __forceinline__ void wait_async() {
#ifdef ASYNC_LDS
#if __has_builtin(__builtin_amdgcn_s_wait_asynccnt)
  __builtin_amdgcn_s_wait_asynccnt(N);
#else
  asm volatile("s_wait_asynccnt %0" ::"i"(N) : "memory");
#endif
#endif
}

// ---------------------------------------------------------------------------
// Elementwise f32 -> bf16 conversion (one-time per operand).
// ---------------------------------------------------------------------------
__global__ __launch_bounds__(256)
void to_bf16(const float* __restrict__ s, unsigned short* __restrict__ d, long n)
{
  long i = ((long)blockIdx.x * blockDim.x + threadIdx.x) * 4;
  if (i + 3 < n) {
    const float4 f = *reinterpret_cast<const float4*>(s + i);
    uint2 p; p.x = packbf2(f.x, f.y); p.y = packbf2(f.z, f.w);
    *reinterpret_cast<uint2*>(d + i) = p;
  }
}

// ---------------------------------------------------------------------------
// GEMM: Y[M,N] = act(A[M,K] @ W[N,K]^T + bias[N]) * outScale
// A, W are bf16. Tile 128x128x32, double-buffered async K-tiles.
// 256 threads = 8 waves in 4(m) x 2(n); each wave: 2x4 WMMA tiles.
// ---------------------------------------------------------------------------
template <bool GELU, bool OUT_BF16>
__global__ __launch_bounds__(256)
void gemm_bf16(const unsigned short* __restrict__ A,
               const unsigned short* __restrict__ Wt,
               const float* __restrict__ bias, void* __restrict__ Yout,
               int M, int N, int K, float outScale)
{
  __shared__ __align__(16) unsigned short As[2][128][32];  // [buf][m][k]
  __shared__ __align__(16) unsigned short Bs[2][128][32];  // [buf][n][k]

  const int tid   = threadIdx.x;
  const int lane  = tid & 31;
  const int wave  = tid >> 5;
  const int wm    = wave & 3;
  const int wn    = wave >> 2;
  const int lmod  = lane & 15;
  const int lhalf = lane >> 4;
  const int row0  = blockIdx.y * 128;
  const int col0  = blockIdx.x * 128;

  v8f acc[2][4] = {};

  auto stage = [&](int kt, int buf) {
#pragma unroll
    for (int i = 0; i < 2; ++i) {
      int c  = tid + (i << 8);            // 512 x b128 chunks per tile
      int r  = c >> 2;
      int c8 = (c & 3) << 3;
      cp_b128(A  + (size_t)(row0 + r) * K + (size_t)kt * 32 + c8, &As[buf][r][c8]);
      cp_b128(Wt + (size_t)(col0 + r) * K + (size_t)kt * 32 + c8, &Bs[buf][r][c8]);
    }
  };

  const int KT = K >> 5;
  stage(0, 0);
  for (int kt = 0; kt < KT; ++kt) {
    const int cur = kt & 1;
    if (kt + 1 < KT) {
      stage(kt + 1, cur ^ 1);   // prefetch next tile into other buffer
      wait_async<4>();          // allow the 4 just-issued async ops in flight
    } else {
      wait_async<0>();
    }
    __syncthreads();

    Frag bfr[4];
#pragma unroll
    for (int tn = 0; tn < 4; ++tn) {
      int n = wn * 64 + tn * 16 + lmod;   // B 32x16: lane = column, K pairs
      int koff = lhalf * 16;
#pragma unroll
      for (int j = 0; j < 8; ++j)
        bfr[tn].u[j] =
            *reinterpret_cast<const unsigned int*>(&Bs[cur][n][koff + 2 * j]);
    }
    Frag afr[2];
#pragma unroll
    for (int tm = 0; tm < 2; ++tm) {
      int r = wm * 32 + tm * 16 + lmod;   // A 16x32: lane = row
#pragma unroll
      for (int j = 0; j < 8; ++j)
        afr[tm].u[j] =
            *reinterpret_cast<const unsigned int*>(&As[cur][r][a_kd(j, lhalf)]);
    }

#pragma unroll
    for (int tm = 0; tm < 2; ++tm)
#pragma unroll
      for (int tn = 0; tn < 4; ++tn)
        acc[tm][tn] = __builtin_amdgcn_wmma_f32_16x16x32_bf16(
            false, afr[tm].v, false, bfr[tn].v, (short)0, acc[tm][tn],
            false, false);
    __syncthreads();
  }

  // ---- epilogue ----
#pragma unroll
  for (int tm = 0; tm < 2; ++tm) {
#pragma unroll
    for (int tn = 0; tn < 4; ++tn) {
      int col = col0 + wn * 64 + tn * 16 + lmod;
      float bv = bias[col];
#pragma unroll
      for (int r = 0; r < 8; ++r) {
        int row = row0 + wm * 32 + tm * 16 + r + 8 * lhalf;
        float vv = acc[tm][tn][r] + bv;
        if (GELU) vv = 0.5f * vv * (1.0f + erff(vv * 0.70710678118654752f));
        vv *= outScale;
        if (OUT_BF16)
          ((unsigned short*)Yout)[(size_t)row * N + col] = f2bf(vv);
        else
          ((float*)Yout)[(size_t)row * N + col] = vv;
      }
    }
  }
}

// ---------------------------------------------------------------------------
// Flash attention on bf16 Q/K/V (Q pre-scaled by 1/sqrt(64) in its GEMM).
// One workgroup = (b, h, 128 queries); 8 waves, one 16-query tile each.
// 64-key chunks, K and V async-copied to LDS (double-buffered), online
// softmax in f32 (one reduction round per 64 keys), P bounced through
// per-wave LDS, PV B-fragments gathered as ushort pairs. bf16 output.
// ---------------------------------------------------------------------------
__global__ __launch_bounds__(256)
void flash_attn(const unsigned short* __restrict__ q,
                const unsigned short* __restrict__ k,
                const unsigned short* __restrict__ v,
                unsigned short* __restrict__ o,
                int B, int S, int H, int D)
{
  __shared__ __align__(16) unsigned short Ks[2][64][64];   // [buf][key][d]
  __shared__ __align__(16) unsigned short Vs[2][64][64];   // [buf][key][d]
  __shared__ __align__(16) unsigned short Ps[8][16][64];   // per-wave P tile

  const int tid   = threadIdx.x;
  const int lane  = tid & 31;
  const int wave  = tid >> 5;
  const int lmod  = lane & 15;
  const int lhalf = lane >> 4;

  const int b = blockIdx.z;
  const int h = blockIdx.y;
  const int q_row0 = blockIdx.x * 128 + wave * 16;
  const size_t base = ((size_t)b * S) * D + h * HEAD_DIM;

  // ---- Q fragments (2 halves of d=64) straight from bf16 global ----
  Frag qf[2];
#pragma unroll
  for (int dh = 0; dh < 2; ++dh) {
    const unsigned short* qrow = q + base + (size_t)(q_row0 + lmod) * D + dh * 32;
#pragma unroll
    for (int j = 0; j < 8; ++j)
      qf[dh].u[j] =
          *reinterpret_cast<const unsigned int*>(qrow + a_kd(j, lhalf));
  }

  float mrow[8], srow[8];
#pragma unroll
  for (int r = 0; r < 8; ++r) { mrow[r] = -3.0e38f; srow[r] = 0.0f; }
  v8f oa[4] = {};

  auto stageKV = [&](int kc, int buf) {
    const int key0 = kc * 64;
#pragma unroll
    for (int i = 0; i < 2; ++i) {
      int c  = tid + (i << 8);            // 512 x b128 chunks per matrix
      int kk = c >> 3;
      int d8 = (c & 7) << 3;
      cp_b128(k + base + (size_t)(key0 + kk) * D + d8, &Ks[buf][kk][d8]);
      cp_b128(v + base + (size_t)(key0 + kk) * D + d8, &Vs[buf][kk][d8]);
    }
  };

  const int nchunks = S / 64;
  stageKV(0, 0);
  for (int kc = 0; kc < nchunks; ++kc) {
    const int cur = kc & 1;
    if (kc + 1 < nchunks) {
      stageKV(kc + 1, cur ^ 1);
      wait_async<4>();
    } else {
      wait_async<0>();
    }
    __syncthreads();

    // ---- scores: 16 queries x 64 keys = 4 WMMA tiles over d=64 ----
    v8f sc[4] = {};
#pragma unroll
    for (int nt = 0; nt < 4; ++nt) {
#pragma unroll
      for (int dh = 0; dh < 2; ++dh) {
        Frag kb;
#pragma unroll
        for (int j = 0; j < 8; ++j)
          kb.u[j] = *reinterpret_cast<const unsigned int*>(
              &Ks[cur][nt * 16 + lmod][dh * 32 + lhalf * 16 + 2 * j]);
        sc[nt] = __builtin_amdgcn_wmma_f32_16x16x32_bf16(
            false, qf[dh].v, false, kb.v, (short)0, sc[nt], false, false);
      }
    }

    // ---- online softmax (row = r + 8*lhalf, N striped over 16 lanes) ----
    float corr[8];
#pragma unroll
    for (int r = 0; r < 8; ++r) {
      float mx = fmaxf(fmaxf(sc[0][r], sc[1][r]), fmaxf(sc[2][r], sc[3][r]));
#pragma unroll
      for (int m = 1; m < 16; m <<= 1) mx = fmaxf(mx, __shfl_xor(mx, m, 32));
      float mnew = fmaxf(mrow[r], mx);
      float p0 = __expf(sc[0][r] - mnew);
      float p1 = __expf(sc[1][r] - mnew);
      float p2 = __expf(sc[2][r] - mnew);
      float p3 = __expf(sc[3][r] - mnew);
      sc[0][r] = p0; sc[1][r] = p1; sc[2][r] = p2; sc[3][r] = p3;
      float rs = (p0 + p1) + (p2 + p3);
#pragma unroll
      for (int m = 1; m < 16; m <<= 1) rs += __shfl_xor(rs, m, 32);
      float c = __expf(mrow[r] - mnew);
      srow[r] = srow[r] * c + rs;
      mrow[r] = mnew;
      corr[r] = c;
    }
#pragma unroll
    for (int t = 0; t < 4; ++t)
#pragma unroll
      for (int r = 0; r < 8; ++r) oa[t][r] *= corr[r];

    // ---- C-layout -> A-layout for P (16x64) via per-wave LDS bounce ----
#pragma unroll
    for (int nt = 0; nt < 4; ++nt)
#pragma unroll
      for (int r = 0; r < 8; ++r)
        Ps[wave][r + 8 * lhalf][nt * 16 + lmod] = f2bf(sc[nt][r]);
    asm volatile("s_wait_dscnt 0" ::: "memory");

    Frag pf[2];
#pragma unroll
    for (int pi = 0; pi < 2; ++pi)
#pragma unroll
      for (int j = 0; j < 8; ++j)
        pf[pi].u[j] = *reinterpret_cast<const unsigned int*>(
            &Ps[wave][lmod][pi * 32 + a_kd(j, lhalf)]);

    // ---- O += P(16x64) @ V(64x64): 2x4 WMMA tiles ----
#pragma unroll
    for (int pi = 0; pi < 2; ++pi) {
#pragma unroll
      for (int t = 0; t < 4; ++t) {
        Frag vb;
#pragma unroll
        for (int j = 0; j < 8; ++j) {
          int kk = pi * 32 + lhalf * 16 + 2 * j;   // key index pair
          vb.s[2 * j]     = Vs[cur][kk][t * 16 + lmod];
          vb.s[2 * j + 1] = Vs[cur][kk + 1][t * 16 + lmod];
        }
        oa[t] = __builtin_amdgcn_wmma_f32_16x16x32_bf16(
            false, pf[pi].v, false, vb.v, (short)0, oa[t], false, false);
      }
    }
    __syncthreads();
  }

  // ---- normalize and store bf16 O ----
  float inv[8];
#pragma unroll
  for (int r = 0; r < 8; ++r) inv[r] = 1.0f / srow[r];
#pragma unroll
  for (int t = 0; t < 4; ++t)
#pragma unroll
    for (int r = 0; r < 8; ++r) {
      int qr = q_row0 + r + 8 * lhalf;
      o[base + (size_t)qr * D + t * 16 + lmod] = f2bf(oa[t][r] * inv[r]);
    }
}

// ---------------------------------------------------------------------------
// Fused residual-add + LayerNorm over D_MODEL=1024 (one block per row).
// Optionally emits a bf16 copy for the downstream GEMM.
// ---------------------------------------------------------------------------
template <bool EMIT_BF16>
__global__ __launch_bounds__(256)
void add_layernorm(const float* __restrict__ x, const float* __restrict__ resid,
                   const float* __restrict__ g, const float* __restrict__ beta,
                   float* __restrict__ y, unsigned short* __restrict__ yb)
{
  const int row = blockIdx.x;
  const int tid = threadIdx.x;
  const float4 a = *reinterpret_cast<const float4*>(x + (size_t)row * D_MODEL + tid * 4);
  const float4 b = *reinterpret_cast<const float4*>(resid + (size_t)row * D_MODEL + tid * 4);
  float4 t;
  t.x = a.x + b.x; t.y = a.y + b.y; t.z = a.z + b.z; t.w = a.w + b.w;
  float s  = t.x + t.y + t.z + t.w;
  float ss = t.x * t.x + t.y * t.y + t.z * t.z + t.w * t.w;
#pragma unroll
  for (int m = 1; m < 32; m <<= 1) {
    s  += __shfl_xor(s, m, 32);
    ss += __shfl_xor(ss, m, 32);
  }
  __shared__ float ps[8], pss[8];
  __shared__ float mu_s, rs_s;
  if ((tid & 31) == 0) { ps[tid >> 5] = s; pss[tid >> 5] = ss; }
  __syncthreads();
  if (tid == 0) {
    float S = 0.f, SS = 0.f;
#pragma unroll
    for (int i = 0; i < 8; ++i) { S += ps[i]; SS += pss[i]; }
    float mu  = S * (1.0f / D_MODEL);
    float var = SS * (1.0f / D_MODEL) - mu * mu;
    mu_s = mu;
    rs_s = rsqrtf(var + 1e-5f);
  }
  __syncthreads();
  const float mu = mu_s, rs = rs_s;
  const int c = tid * 4;
  const float4 gg = *reinterpret_cast<const float4*>(g + c);
  const float4 bb = *reinterpret_cast<const float4*>(beta + c);
  float4 out;
  out.x = (t.x - mu) * rs * gg.x + bb.x;
  out.y = (t.y - mu) * rs * gg.y + bb.y;
  out.z = (t.z - mu) * rs * gg.z + bb.z;
  out.w = (t.w - mu) * rs * gg.w + bb.w;
  *reinterpret_cast<float4*>(y + (size_t)row * D_MODEL + c) = out;
  if (EMIT_BF16) {
    uint2 p; p.x = packbf2(out.x, out.y); p.y = packbf2(out.z, out.w);
    *reinterpret_cast<uint2*>(yb + (size_t)row * D_MODEL + c) = p;
  }
}

// ---------------------------------------------------------------------------
// Orchestration. Workspace (~248 MB): bf16 copies of x + 6 weights, bf16
// Q/K/V/O/H activations, f32 proj/X1 for residual paths.
// ---------------------------------------------------------------------------
extern "C" void kernel_launch(void* const* d_in, const int* in_sizes, int n_in,
                              void* d_out, int out_size, void* d_ws, size_t ws_size,
                              hipStream_t stream)
{
  (void)in_sizes; (void)n_in; (void)out_size; (void)ws_size;
  const float* x     = (const float*)d_in[0];
  const float* wq    = (const float*)d_in[1];
  const float* bq    = (const float*)d_in[2];
  const float* wk    = (const float*)d_in[3];
  const float* bk    = (const float*)d_in[4];
  const float* wv    = (const float*)d_in[5];
  const float* bv    = (const float*)d_in[6];
  const float* wo    = (const float*)d_in[7];
  const float* bo    = (const float*)d_in[8];
  const float* w1    = (const float*)d_in[9];
  const float* b1    = (const float*)d_in[10];
  const float* w2    = (const float*)d_in[11];
  const float* b2    = (const float*)d_in[12];
  const float* g1    = (const float*)d_in[13];
  const float* beta1 = (const float*)d_in[14];
  const float* g2    = (const float*)d_in[15];
  const float* beta2 = (const float*)d_in[16];

  char* p = (char*)d_ws;
  auto alloc = [&](size_t bytes) {
    void* r = (void*)p;
    p += (bytes + 255) & ~(size_t)255;
    return r;
  };
  const size_t TOK = (size_t)ROWS * D_MODEL;            // 8M elements
  const size_t WDD = (size_t)D_MODEL * D_MODEL;         // 1M
  const size_t WFD = (size_t)FF_DIM * D_MODEL;          // 4M

  unsigned short* xb  = (unsigned short*)alloc(TOK * 2);
  unsigned short* wqb = (unsigned short*)alloc(WDD * 2);
  unsigned short* wkb = (unsigned short*)alloc(WDD * 2);
  unsigned short* wvb = (unsigned short*)alloc(WDD * 2);
  unsigned short* wob = (unsigned short*)alloc(WDD * 2);
  unsigned short* w1b = (unsigned short*)alloc(WFD * 2);
  unsigned short* w2b = (unsigned short*)alloc(WFD * 2);
  unsigned short* Qb  = (unsigned short*)alloc(TOK * 2);
  unsigned short* Kb  = (unsigned short*)alloc(TOK * 2);
  unsigned short* Vb  = (unsigned short*)alloc(TOK * 2);
  unsigned short* Ob  = (unsigned short*)alloc(TOK * 2);
  unsigned short* Hb  = (unsigned short*)alloc((size_t)ROWS * FF_DIM * 2);
  unsigned short* X1b = (unsigned short*)alloc(TOK * 2);
  float*          X1  = (float*)alloc(TOK * 4);
  float*          Prj = (float*)alloc(TOK * 4);         // reused for FF out

  const dim3 blk(256);
  const dim3 gD(D_MODEL / 128, ROWS / 128);             // (8, 64)
  const dim3 gF(FF_DIM / 128, ROWS / 128);              // (32, 64)

  // one-time bf16 conversions (~21M elements total, bandwidth-trivial)
  to_bf16<<<(unsigned)(TOK / 4 / 256), blk, 0, stream>>>(x,  xb,  (long)TOK);
  to_bf16<<<(unsigned)(WDD / 4 / 256), blk, 0, stream>>>(wq, wqb, (long)WDD);
  to_bf16<<<(unsigned)(WDD / 4 / 256), blk, 0, stream>>>(wk, wkb, (long)WDD);
  to_bf16<<<(unsigned)(WDD / 4 / 256), blk, 0, stream>>>(wv, wvb, (long)WDD);
  to_bf16<<<(unsigned)(WDD / 4 / 256), blk, 0, stream>>>(wo, wob, (long)WDD);
  to_bf16<<<(unsigned)(WFD / 4 / 256), blk, 0, stream>>>(w1, w1b, (long)WFD);
  to_bf16<<<(unsigned)(WFD / 4 / 256), blk, 0, stream>>>(w2, w2b, (long)WFD);

  // QKV projections (Q pre-scaled by 1/sqrt(head_dim))
  gemm_bf16<false, true><<<gD, blk, 0, stream>>>(xb, wqb, bq, Qb, ROWS, D_MODEL, D_MODEL, 0.125f);
  gemm_bf16<false, true><<<gD, blk, 0, stream>>>(xb, wkb, bk, Kb, ROWS, D_MODEL, D_MODEL, 1.0f);
  gemm_bf16<false, true><<<gD, blk, 0, stream>>>(xb, wvb, bv, Vb, ROWS, D_MODEL, D_MODEL, 1.0f);

  // attention
  flash_attn<<<dim3(SEQ / 128, N_HEADS, BATCH), blk, 0, stream>>>(
      Qb, Kb, Vb, Ob, BATCH, SEQ, N_HEADS, D_MODEL);

  // output projection -> f32, then residual + LN1 (emits f32 + bf16)
  gemm_bf16<false, false><<<gD, blk, 0, stream>>>(Ob, wob, bo, Prj, ROWS, D_MODEL, D_MODEL, 1.0f);
  add_layernorm<true><<<ROWS, blk, 0, stream>>>(x, Prj, g1, beta1, X1, X1b);

  // FFN
  gemm_bf16<true, true><<<gF, blk, 0, stream>>>(X1b, w1b, b1, Hb, ROWS, FF_DIM, D_MODEL, 1.0f);
  gemm_bf16<false, false><<<gD, blk, 0, stream>>>(Hb, w2b, b2, Prj, ROWS, D_MODEL, FF_DIM, 1.0f);
  add_layernorm<false><<<ROWS, blk, 0, stream>>>(X1, Prj, g2, beta2, (float*)d_out, nullptr);
}
